// SDCNv3_2774548873826
// MI455X (gfx1250) — matrix operand order repaired and checked by reading, not compiled, not run on hardware
//
#include <hip/hip_runtime.h>

// ---------------------------------------------------------------------------
// Sizes
// ---------------------------------------------------------------------------
#define BATCH      8192
#define EMBED_DIM  256
#define INPUT_DIM  768
#define NUM_LAYERS 4
#define NELEM      (BATCH * INPUT_DIM)        // 6291456
#define W_ELEMS    (NUM_LAYERS * INPUT_DIM * INPUT_DIM) // 2359296
#define EPS        1e-5f

typedef __attribute__((ext_vector_type(16))) __bf16 bf16x16;
typedef __attribute__((ext_vector_type(8)))  float  f32x8;

struct __align__(16) U4 { unsigned int a, b, c, d; };
union FragU { bf16x16 v; U4 q[2]; };

__device__ __forceinline__ unsigned short f2bf(float f) {
  unsigned int u = __float_as_uint(f);
  unsigned int r = 0x7FFFu + ((u >> 16) & 1u);   // round-to-nearest-even
  return (unsigned short)((u + r) >> 16);
}

__device__ __forceinline__ float sigmoidf_(float z) {
  return 1.0f / (1.0f + expf(-z));
}

// Threefry-2x32, 20 rounds (JAX-compatible structure)
__device__ __forceinline__ void threefry2x32(unsigned int k0, unsigned int k1,
                                             unsigned int x0, unsigned int x1,
                                             unsigned int& o0, unsigned int& o1) {
  const unsigned int ks2 = 0x1BD11BDAu ^ k0 ^ k1;
  unsigned int y0 = x0 + k0, y1 = x1 + k1;
  const int rA[4] = {13, 15, 26, 6};
  const int rB[4] = {17, 29, 16, 24};
#define TF_ROT(x, n) (((x) << (n)) | ((x) >> (32 - (n))))
#pragma unroll
  for (int i = 0; i < 4; ++i) { y0 += y1; y1 = TF_ROT(y1, rA[i]); y1 ^= y0; }
  y0 += k1;  y1 += ks2 + 1u;
#pragma unroll
  for (int i = 0; i < 4; ++i) { y0 += y1; y1 = TF_ROT(y1, rB[i]); y1 ^= y0; }
  y0 += ks2; y1 += k0 + 2u;
#pragma unroll
  for (int i = 0; i < 4; ++i) { y0 += y1; y1 = TF_ROT(y1, rA[i]); y1 ^= y0; }
  y0 += k0;  y1 += k1 + 3u;
#pragma unroll
  for (int i = 0; i < 4; ++i) { y0 += y1; y1 = TF_ROT(y1, rB[i]); y1 ^= y0; }
  y0 += k1;  y1 += ks2 + 4u;
#pragma unroll
  for (int i = 0; i < 4; ++i) { y0 += y1; y1 = TF_ROT(y1, rA[i]); y1 ^= y0; }
  y0 += ks2; y1 += k0 + 5u;
#undef TF_ROT
  o0 = y0; o1 = y1;
}

// ---------------------------------------------------------------------------
// f32 -> bf16 bulk convert (weights)
// ---------------------------------------------------------------------------
__global__ void cvt_bf16_kernel(const float* __restrict__ in,
                                unsigned short* __restrict__ out, int n) {
  int i = blockIdx.x * blockDim.x + threadIdx.x;
  if (i < n) out[i] = f2bf(in[i]);
}

// ---------------------------------------------------------------------------
// Embedding gather + half-split concat: builds x (f32 twice, bf16 twice)
// ---------------------------------------------------------------------------
__global__ void embed_kernel(const int* __restrict__ xp, const int* __restrict__ xa,
                             const int* __restrict__ xc,
                             const float* __restrict__ E0, const float* __restrict__ E1,
                             const float* __restrict__ E2,
                             float* __restrict__ xd, float* __restrict__ xs,
                             unsigned short* __restrict__ xdb,
                             unsigned short* __restrict__ xsb) {
  int i = blockIdx.x * 256 + threadIdx.x;
  if (i >= NELEM) return;
  int b = i / INPUT_DIM;
  int j = i - b * INPUT_DIM;
  int seg = j >> 7;          // 6 segments of 128
  int off = j & 127;
  float v;
  if (seg == 0)      v = E0[(size_t)xp[b] * EMBED_DIM + off];
  else if (seg == 1) v = E1[(size_t)xa[b] * EMBED_DIM + off];
  else if (seg == 2) v = E2[(size_t)xc[b] * EMBED_DIM + off];
  else if (seg == 3) v = E0[(size_t)xp[b] * EMBED_DIM + 128 + off];
  else if (seg == 4) v = E1[(size_t)xa[b] * EMBED_DIM + 128 + off];
  else               v = E2[(size_t)xc[b] * EMBED_DIM + 128 + off];
  xd[i] = v; xs[i] = v;
  unsigned short h = f2bf(v);
  xdb[i] = h; xsb[i] = h;
}

// ---------------------------------------------------------------------------
// bf16 WMMA GEMM: cross[m,n] = sum_k A[m,k]*W[n,k] + bias[n]
//   block tile 128x128, 8 waves, wave = 2x4 fragments of 16x16, K step 32.
//   Double-buffered LDS fed by GLOBAL_LOAD_ASYNC_TO_LDS_B128 (ASYNCcnt).
//   Also emits per-block partial sums of cross and cross^2.
// ---------------------------------------------------------------------------
#define TM 128
#define TN 128
#define TK 32
#define LDP 40   // padded LDS row (halves): 80B stride -> conflict-free b128 reads

__global__ __launch_bounds__(256) void gemm_cross_kernel(
    const unsigned short* __restrict__ A,    // [BATCH x 768] bf16
    const unsigned short* __restrict__ W,    // [768 x 768] bf16 (layer base)
    const float* __restrict__ bias,          // [768]
    float* __restrict__ crossOut,            // [BATCH x 768] f32
    float* __restrict__ pS1, float* __restrict__ pS2) {
  __shared__ __align__(16) unsigned short As[2][TM][LDP];
  __shared__ __align__(16) unsigned short Bs[2][TN][LDP];
  __shared__ float r1[256], r2[256];

  const int tid   = threadIdx.x;
  const int lane  = tid & 31;
  const int wave  = tid >> 5;
  const int ll    = lane & 15;
  const int hi    = lane >> 4;
  const int waveM = wave & 3;    // 4 groups x 32 rows
  const int waveN = wave >> 2;   // 2 groups x 64 cols
  const int m0 = blockIdx.x * TM;
  const int n0 = blockIdx.y * TN;

  f32x8 acc[8] = {};

  // staging: each thread moves 2x16B of A and 2x16B of W per k-tile
  const int rowS = tid >> 1;
  const int qS   = (tid & 1) * 2;   // chunks qS, qS+1 (8 halves each)

  // async global -> LDS stage of one 128x32 A tile + 128x32 W tile
  auto stage_async = [&](int kb, int buf) {
    const unsigned short* ga0 = &A[(size_t)(m0 + rowS) * INPUT_DIM + kb + (qS + 0) * 8];
    const unsigned short* ga1 = &A[(size_t)(m0 + rowS) * INPUT_DIM + kb + (qS + 1) * 8];
    const unsigned short* gb0 = &W[(size_t)(n0 + rowS) * INPUT_DIM + kb + (qS + 0) * 8];
    const unsigned short* gb1 = &W[(size_t)(n0 + rowS) * INPUT_DIM + kb + (qS + 1) * 8];
    unsigned la0 = (unsigned)(size_t)&As[buf][rowS][(qS + 0) * 8];
    unsigned la1 = (unsigned)(size_t)&As[buf][rowS][(qS + 1) * 8];
    unsigned lb0 = (unsigned)(size_t)&Bs[buf][rowS][(qS + 0) * 8];
    unsigned lb1 = (unsigned)(size_t)&Bs[buf][rowS][(qS + 1) * 8];
    asm volatile("global_load_async_to_lds_b128 %0, %1, off" :: "v"(la0), "v"(ga0) : "memory");
    asm volatile("global_load_async_to_lds_b128 %0, %1, off" :: "v"(la1), "v"(ga1) : "memory");
    asm volatile("global_load_async_to_lds_b128 %0, %1, off" :: "v"(lb0), "v"(gb0) : "memory");
    asm volatile("global_load_async_to_lds_b128 %0, %1, off" :: "v"(lb1), "v"(gb1) : "memory");
  };

  stage_async(0, 0);
  asm volatile("s_wait_asynccnt 0x0" ::: "memory");
  __syncthreads();

  int buf = 0;
  for (int kb = 0; kb < INPUT_DIM; kb += TK) {
    if (kb + TK < INPUT_DIM) stage_async(kb + TK, buf ^ 1);

    // A fragment (16-bit A 16x32): half-wave hi: halves 0..7 = K hi*8..,
    // halves 8..15 = K 16+hi*8..  -> two b128 LDS reads.
    FragU af[2];
#pragma unroll
    for (int fm = 0; fm < 2; ++fm) {
      int r = waveM * 32 + fm * 16 + ll;
      af[fm].q[0] = *reinterpret_cast<const U4*>(&As[buf][r][hi * 8]);
      af[fm].q[1] = *reinterpret_cast<const U4*>(&As[buf][r][16 + hi * 8]);
    }
    // B fragment (32x16): column n = W row n; half-wave hi: halves 0..7 = K hi*8..,
    // halves 8..15 = K 16+hi*8.. (same order as A).
    FragU bfg[4];
#pragma unroll
    for (int fn = 0; fn < 4; ++fn) {
      int r = waveN * 64 + fn * 16 + ll;
      bfg[fn].q[0] = *reinterpret_cast<const U4*>(&Bs[buf][r][hi * 8]);
      bfg[fn].q[1] = *reinterpret_cast<const U4*>(&Bs[buf][r][16 + hi * 8]);
    }
#pragma unroll
    for (int fm = 0; fm < 2; ++fm)
#pragma unroll
      for (int fn = 0; fn < 4; ++fn)
        acc[fm * 4 + fn] = __builtin_amdgcn_wmma_f32_16x16x32_bf16(
            false, af[fm].v, false, bfg[fn].v, (short)0, acc[fm * 4 + fn],
            false, false);

    asm volatile("s_wait_asynccnt 0x0" ::: "memory");
    __syncthreads();
    buf ^= 1;
  }

  // Epilogue: bias add, store cross, accumulate block partials of c and c^2.
  float s1 = 0.0f, s2 = 0.0f;
#pragma unroll
  for (int fm = 0; fm < 2; ++fm)
#pragma unroll
    for (int fn = 0; fn < 4; ++fn) {
      int gnb = n0 + waveN * 64 + fn * 16 + ll;
      float bv = bias[gnb];
#pragma unroll
      for (int v = 0; v < 8; ++v) {
        int gm = m0 + waveM * 32 + fm * 16 + hi * 8 + v;  // C layout: M = hi*8+v
        float c = acc[fm * 4 + fn][v] + bv;
        crossOut[(size_t)gm * INPUT_DIM + gnb] = c;
        s1 += c; s2 += c * c;
      }
    }
  r1[tid] = s1; r2[tid] = s2;
  __syncthreads();
  for (int s = 128; s > 0; s >>= 1) {
    if (tid < s) { r1[tid] += r1[tid + s]; r2[tid] += r2[tid + s]; }
    __syncthreads();
  }
  if (tid == 0) {
    int bi = blockIdx.y * gridDim.x + blockIdx.x;
    pS1[bi] = r1[0]; pS2[bi] = r2[0];
  }
}

// ---------------------------------------------------------------------------
// Stats: mean + rsqrt(var+eps) over full tensor; fold threefry key for layer.
// ---------------------------------------------------------------------------
__global__ void stats_kernel(const float* __restrict__ pS1,
                             const float* __restrict__ pS2,
                             float* __restrict__ stats, int nparts, int step_id) {
  __shared__ float r1[256], r2[256];
  float s1 = 0.0f, s2 = 0.0f;
  for (int i = threadIdx.x; i < nparts; i += 256) { s1 += pS1[i]; s2 += pS2[i]; }
  r1[threadIdx.x] = s1; r2[threadIdx.x] = s2;
  __syncthreads();
  for (int s = 128; s > 0; s >>= 1) {
    if (threadIdx.x < s) { r1[threadIdx.x] += r1[threadIdx.x + s];
                           r2[threadIdx.x] += r2[threadIdx.x + s]; }
    __syncthreads();
  }
  if (threadIdx.x == 0) {
    const float invN = 1.0f / (float)NELEM;
    float m = r1[0] * invN;
    float v = r2[0] * invN - m * m;
    stats[0] = m;
    stats[1] = rsqrtf(v + EPS);
    unsigned int o0, o1;                       // fold_in(key(42), step_id)
    threefry2x32(0u, 42u, 0u, (unsigned int)step_id, o0, o1);
    stats[2] = __uint_as_float(o0);
    stats[3] = __uint_as_float(o1);
  }
}

// ---------------------------------------------------------------------------
// Update: ln -> sigmoid -> threefry bernoulli mask -> x += cross*mask; re-emit bf16
// ---------------------------------------------------------------------------
__global__ void update_kernel(const float* __restrict__ cross,
                              const float* __restrict__ stats,
                              float* __restrict__ x,
                              unsigned short* __restrict__ xbf) {
  int i = blockIdx.x * 256 + threadIdx.x;
  if (i >= NELEM) return;
  const float m = stats[0], rstd = stats[1];
  const unsigned int k0 = __float_as_uint(stats[2]);
  const unsigned int k1 = __float_as_uint(stats[3]);
  float c = cross[i];
  float p = sigmoidf_((c - m) * rstd);
  const int half = NELEM >> 1;                 // counter split i <-> i+half
  unsigned int pi = (i < half) ? (unsigned int)i : (unsigned int)(i - half);
  unsigned int o0, o1;
  threefry2x32(k0, k1, pi, pi + (unsigned int)half, o0, o1);
  unsigned int bits = (i < half) ? o0 : o1;
  float u = __uint_as_float(0x3f800000u | (bits >> 9)) - 1.0f;  // [0,1)
  float xn = x[i] + ((u < p) ? c : 0.0f);
  x[i] = xn;
  xbf[i] = f2bf(xn);
}

// ---------------------------------------------------------------------------
// Heads: 4 dot products per row, sigmoids, y_pred_d out, per-block BCE partials
// ---------------------------------------------------------------------------
__global__ __launch_bounds__(256) void head_kernel(
    const float* __restrict__ xd, const float* __restrict__ xs,
    const float* __restrict__ wfd, const float* __restrict__ bfd,
    const float* __restrict__ wfs, const float* __restrict__ bfs,
    const float* __restrict__ y_true, float* __restrict__ out,
    float* __restrict__ lpd, float* __restrict__ lps) {
  const int wave = threadIdx.x >> 5, lane = threadIdx.x & 31;
  const int b = blockIdx.x * 8 + wave;
  float dd = 0.f, ds = 0.f, sd = 0.f, ss = 0.f;
  for (int j = lane; j < INPUT_DIM; j += 32) {
    float vd = xd[(size_t)b * INPUT_DIM + j];
    float vs = xs[(size_t)b * INPUT_DIM + j];
    float wd = wfd[j], wv = wfs[j];
    dd += vd * wd; ds += vd * wv; sd += vs * wd; ss += vs * wv;
  }
  for (int off = 16; off > 0; off >>= 1) {
    dd += __shfl_down(dd, off, 32);
    ds += __shfl_down(ds, off, 32);
    sd += __shfl_down(sd, off, 32);
    ss += __shfl_down(ss, off, 32);
  }
  __shared__ float shD[8], shS[8];
  if (lane == 0) {
    float sd_d = sigmoidf_(dd + bfd[0]);
    float ss_s = sigmoidf_(ss + bfs[0]);
    float sd_s = sigmoidf_(sd + bfd[0]);
    float ss_d = sigmoidf_(ds + bfs[0]);
    float ypd = 0.5f * (sd_d + ss_s);
    float yps = 0.5f * (sd_s + ss_d);
    out[b] = ypd;
    float y = y_true[b];
    shD[wave] = -(y * logf(ypd) + (1.f - y) * logf(1.f - ypd));
    shS[wave] = -(y * logf(yps) + (1.f - y) * logf(1.f - yps));
  }
  __syncthreads();
  if (threadIdx.x == 0) {
    float t1 = 0.f, t2 = 0.f;
    for (int i = 0; i < 8; ++i) { t1 += shD[i]; t2 += shS[i]; }
    lpd[blockIdx.x] = t1; lps[blockIdx.x] = t2;
  }
}

__global__ void final_loss_kernel(const float* __restrict__ lpd,
                                  const float* __restrict__ lps,
                                  int n, float* __restrict__ out) {
  __shared__ float r1[256], r2[256];
  float s1 = 0.f, s2 = 0.f;
  for (int i = threadIdx.x; i < n; i += 256) { s1 += lpd[i]; s2 += lps[i]; }
  r1[threadIdx.x] = s1; r2[threadIdx.x] = s2;
  __syncthreads();
  for (int s = 128; s > 0; s >>= 1) {
    if (threadIdx.x < s) { r1[threadIdx.x] += r1[threadIdx.x + s];
                           r2[threadIdx.x] += r2[threadIdx.x + s]; }
    __syncthreads();
  }
  if (threadIdx.x == 0) {
    float ld = r1[0] / (float)BATCH, ls = r2[0] / (float)BATCH;
    float bce = ld + ls;
    float wd = fmaxf(0.f, ld - bce);
    float wv = fmaxf(0.f, ls - bce);
    out[BATCH] = bce + wd * ld + wv * ls;
  }
}

// ---------------------------------------------------------------------------
// Host driver
// ---------------------------------------------------------------------------
extern "C" void kernel_launch(void* const* d_in, const int* in_sizes, int n_in,
                              void* d_out, int out_size, void* d_ws, size_t ws_size,
                              hipStream_t stream) {
  (void)in_sizes; (void)n_in; (void)out_size; (void)ws_size;
  const int*   x_p   = (const int*)d_in[0];
  const int*   x_a   = (const int*)d_in[1];
  const int*   x_c   = (const int*)d_in[2];
  const float* y_tru = (const float*)d_in[3];
  const float* E0    = (const float*)d_in[4];
  const float* E1    = (const float*)d_in[5];
  const float* E2    = (const float*)d_in[6];
  const float* Wd    = (const float*)d_in[7];
  const float* bd    = (const float*)d_in[8];
  const float* Ws    = (const float*)d_in[9];
  const float* bs    = (const float*)d_in[10];
  const float* wfd   = (const float*)d_in[11];
  const float* bfd   = (const float*)d_in[12];
  const float* wfs   = (const float*)d_in[13];
  const float* bfs   = (const float*)d_in[14];
  float* out = (float*)d_out;

  // workspace carve-out (256B-aligned)
  char* base = (char*)d_ws;
  size_t off = 0;
  auto take = [&](size_t bytes) -> void* {
    void* p = base + off;
    off += (bytes + 255) & ~(size_t)255;
    return p;
  };
  float*          XD  = (float*)take((size_t)NELEM * 4);
  float*          XS  = (float*)take((size_t)NELEM * 4);
  float*          CR  = (float*)take((size_t)NELEM * 4);
  unsigned short* XDB = (unsigned short*)take((size_t)NELEM * 2);
  unsigned short* XSB = (unsigned short*)take((size_t)NELEM * 2);
  unsigned short* WDB = (unsigned short*)take((size_t)W_ELEMS * 2);
  unsigned short* WSB = (unsigned short*)take((size_t)W_ELEMS * 2);
  float*          PS1 = (float*)take(384 * 4);
  float*          PS2 = (float*)take(384 * 4);
  float*          ST  = (float*)take(64);
  float*          LPD = (float*)take(1024 * 4);
  float*          LPS = (float*)take(1024 * 4);

  const int nElemBlocks = NELEM / 256;              // 24576 (exact)
  const int cvtBlocks   = (W_ELEMS + 255) / 256;

  cvt_bf16_kernel<<<cvtBlocks, 256, 0, stream>>>(Wd, WDB, W_ELEMS);
  cvt_bf16_kernel<<<cvtBlocks, 256, 0, stream>>>(Ws, WSB, W_ELEMS);
  embed_kernel<<<nElemBlocks, 256, 0, stream>>>(x_p, x_a, x_c, E0, E1, E2,
                                                XD, XS, XDB, XSB);

  dim3 gemmGrid(BATCH / TM, INPUT_DIM / TN);        // 64 x 6 = 384 blocks
  for (int net = 0; net < 2; ++net) {
    const unsigned short* Wb = net ? WSB : WDB;
    const float*          bb = net ? bs  : bd;
    float*                X  = net ? XS  : XD;
    unsigned short*       XB = net ? XSB : XDB;
    for (int l = 0; l < NUM_LAYERS; ++l) {
      gemm_cross_kernel<<<gemmGrid, 256, 0, stream>>>(
          XB, Wb + (size_t)l * INPUT_DIM * INPUT_DIM, bb + l * INPUT_DIM,
          CR, PS1, PS2);
      stats_kernel<<<1, 256, 0, stream>>>(PS1, PS2, ST, 384,
                                          net * NUM_LAYERS + l);
      update_kernel<<<nElemBlocks, 256, 0, stream>>>(CR, ST, X, XB);
    }
  }

  head_kernel<<<BATCH / 8, 256, 0, stream>>>(XD, XS, wfd, bfd, wfs, bfs,
                                             y_tru, out, LPD, LPS);
  final_loss_kernel<<<1, 256, 0, stream>>>(LPD, LPS, BATCH / 8, out);
}
